// ScalableGraphSAGE_88373247082993
// MI455X (gfx1250) — matrix operand order repaired and checked by reading, not compiled, not run on hardware
//
#include <hip/hip_runtime.h>
#include <hip/hip_bf16.h>
#include <math.h>

// ---------------------------------------------------------------------------
// GATv2 supply-chain GNN forward for MI455X (gfx1250, wave32, WMMA + TDM).
// GEMMs run on v_wmma_f32_16x16x32_bf16; the 16 x K bf16 weight panel of each
// output-column tile is DMA'd once per workgroup into LDS with
// TENSOR_LOAD_TO_LDS (TENSORcnt + workgroup barrier), B fragments then come
// from ds_load_b128. Irregular segment-softmax uses VALU + float atomics.
// ---------------------------------------------------------------------------

typedef unsigned short u16;
typedef __attribute__((ext_vector_type(16))) __bf16 v16bf;
typedef __attribute__((ext_vector_type(8)))  float  v8f;
typedef __attribute__((ext_vector_type(4)))  unsigned int u32x4;
typedef __attribute__((ext_vector_type(8)))  int i32x8;
typedef __attribute__((ext_vector_type(4)))  int i32x4;

#define HIDC  128
#define HEADS 4
#define CHC   32
#define NEGS  0.2f
#define EPSV  1e-5f
#define KMAX  384   // largest GEMM K (skip projection)

__device__ __forceinline__ u16 f2bf(float f) {
  unsigned u = __float_as_uint(f);
  u += 0x7fffu + ((u >> 16) & 1u);          // round to nearest even
  return (u16)(u >> 16);
}

// float atomic max via signed/unsigned integer atomics (works for all signs,
// init must be -FLT_MAX / -inf)
__device__ __forceinline__ void atomicMaxF(float* addr, float val) {
  if (val >= 0.0f) atomicMax((int*)addr, __float_as_int(val));
  else             atomicMin((unsigned int*)addr, (unsigned int)__float_as_int(val));
}

// ---------------------------------------------------------------------------
// WMMA GEMM:  C[M,N] = A[M,K](bf16) * Bt[N,K](bf16)^T + bias, optional ReLU.
// One wave -> one 16x16 tile, 4 waves per block along M. K % 32 == 0,
// M % 16 == 0, N % 16 == 0 (true for all call sites here).
//
// B panel (Bt rows n0..n0+15, 16*K bf16) is staged into LDS by the Tensor
// Data Mover: 2-D D# descriptor, data_size=2B, tile_dim0=K (contiguous),
// tile_dim1=16, tensor_dim0_stride=K. Wave 0 issues the DMA (TDM ignores
// EXEC), waits TENSORcnt==0, then the block barrier publishes the panel.
//
// Lane layouts per CDNA5 ISA 7.12.2 (wave32):
//   A 16x32 bf16:  row = lane&15, K = 16*(lane>>4)+[0..15] (contiguous order, not the ISA table)
//   B 32x16 bf16:  col = lane&15, K = 16*(lane>>4)+[0..15] (same contiguous order as A, chosen here)
//   C 16x16 f32 :  col = lane&15, rows = 8*(lane>>4) + r, r = 0..7
// ---------------------------------------------------------------------------
__global__ void __launch_bounds__(128) k_gemm_bf16(
    const u16* __restrict__ A, const u16* __restrict__ Bt,
    const float* __restrict__ bias, float* __restrict__ C,
    int M, int N, int K, int relu)
{
  __shared__ __align__(16) u16 Blds[16 * KMAX];
  const int lane = threadIdx.x & 31;
  const int wave = threadIdx.x >> 5;
  const int m0 = (blockIdx.y * 4 + wave) * 16;
  const int n0 = blockIdx.x * 16;

  // ---- TDM: stage 16 x K bf16 weight panel into LDS (wave 0 only) ----
  if (threadIdx.x < 32) {
    const unsigned long long ga = (unsigned long long)(const void*)(Bt + (size_t)n0 * K);
    const unsigned lds_off = (unsigned)(size_t)(void*)&Blds[0];  // addr[31:0] = LDS offset
    u32x4 g0;
    g0[0] = 1u;                                              // count=1, user D#
    g0[1] = lds_off;                                         // lds_addr (bytes)
    g0[2] = (unsigned)(ga & 0xffffffffu);                    // global_addr[31:0]
    g0[3] = (unsigned)((ga >> 32) & 0x01ffffffu) | (2u << 30);  // addr[56:32] | type=2
    i32x8 g1;
    g1[0] = (int)(1u << 16);                                 // data_size=1 -> 2 bytes
    g1[1] = (int)(((unsigned)K & 0xffffu) << 16);            // tensor_dim0[15:0]
    g1[2] = (int)((((unsigned)K >> 16) & 0xffffu)            // tensor_dim0[31:16]
                  | (((unsigned)N & 0xffffu) << 16));        // tensor_dim1[15:0]
    g1[3] = (int)((((unsigned)N >> 16) & 0xffffu)            // tensor_dim1[31:16]
                  | (((unsigned)K & 0xffffu) << 16));        // tile_dim0 = K
    g1[4] = 16;                                              // tile_dim1 = 16 rows
    g1[5] = K;                                               // tensor_dim0_stride
    g1[6] = 0;
    g1[7] = 0;
    i32x4 z4; z4[0] = 0; z4[1] = 0; z4[2] = 0; z4[3] = 0;    // groups 2/3 unused (2-D)
#if __clang_major__ >= 23
    i32x8 z8; z8[0]=0; z8[1]=0; z8[2]=0; z8[3]=0; z8[4]=0; z8[5]=0; z8[6]=0; z8[7]=0;
    __builtin_amdgcn_tensor_load_to_lds(g0, g1, z4, z4, z8, 0);
#else
    __builtin_amdgcn_tensor_load_to_lds(g0, g1, z4, z4, 0);
#endif
    __builtin_amdgcn_s_wait_tensorcnt(0);
  }
  __syncthreads();
  if (m0 >= M) return;

  const int l15 = lane & 15;
  const int hi  = lane >> 4;
  const u16* Ar = A + (size_t)(m0 + l15) * K + hi * 16;    // A fragment base (global)
  const u16* Bl = &Blds[l15 * K + hi * 16];                // B fragment base (LDS)
  v8f acc = {};
  for (int kk = 0; kk < K; kk += 32) {
    union { v16bf v; u16 u[16]; } a, b;
    *(uint4*)&a.u[0] = *(const uint4*)(Ar + kk);        // K = 16*hi + 0..7
    *(uint4*)&a.u[8] = *(const uint4*)(Ar + kk + 8);    // K = 16*hi + 8..15
    *(uint4*)&b.u[0] = *(const uint4*)(Bl + kk);        // K = 16*hi + 0..7
    *(uint4*)&b.u[8] = *(const uint4*)(Bl + kk + 8);    // K = 16*hi + 8..15
    if (kk + 32 < K) __builtin_prefetch(Ar + kk + 32, 0, 3);  // global_prefetch_b8
    acc = __builtin_amdgcn_wmma_f32_16x16x32_bf16(
        false, a.v, false, b.v, (short)0, acc, false, false);
  }
  const int col = n0 + l15;
  const float bv = bias ? bias[col] : 0.0f;
  float* Cp = C + (size_t)(m0 + hi * 8) * N + col;
#pragma unroll
  for (int r = 0; r < 8; ++r) {
    float v = acc[r] + bv;
    if (relu) v = fmaxf(v, 0.0f);
    Cp[(size_t)r * N] = v;
  }
}

// ---- elementwise / conversion helpers -------------------------------------
__global__ void k_fill(float* __restrict__ p, float v, int n) {
  int i = blockIdx.x * blockDim.x + threadIdx.x;
  if (i < n) p[i] = v;
}
__global__ void k_copy(float* __restrict__ d, const float* __restrict__ s, int n) {
  int i = blockIdx.x * blockDim.x + threadIdx.x;
  if (i < n) d[i] = s[i];
}
__global__ void k_conv(const float* __restrict__ s, u16* __restrict__ d, int n) {
  int i = blockIdx.x * blockDim.x + threadIdx.x;
  if (i < n) d[i] = f2bf(s[i]);
}
// W[K,N] fp32 row-major  ->  Bt[N,K] bf16
__global__ void k_tconv(const float* __restrict__ W, u16* __restrict__ Bt, int K, int N) {
  int i = blockIdx.x * blockDim.x + threadIdx.x;
  if (i >= K * N) return;
  int k = i / N, n = i - k * N;
  Bt[(size_t)n * K + k] = f2bf(W[i]);
}
__global__ void k_add(float* __restrict__ a, const float* __restrict__ b, int n) {
  int i = blockIdx.x * blockDim.x + threadIdx.x;
  if (i < n) a[i] += b[i];
}

// ---- node encoder: Linear(8->128)+ReLU -> LayerNorm -> +type embeddings ---
__global__ void k_node_encoder(
    const float* __restrict__ x, const float* __restrict__ W, const float* __restrict__ b,
    const float* __restrict__ lnw, const float* __restrict__ lnb,
    const float* __restrict__ mte, const float* __restrict__ nte,
    const int* __restrict__ mt, const int* __restrict__ nt, float* __restrict__ h)
{
  const int n = blockIdx.x, c = threadIdx.x;      // 128 threads = 4 waves
  const float* xr = x + (size_t)n * 8;
  float acc = b[c];
#pragma unroll
  for (int k = 0; k < 8; ++k) acc = fmaf(xr[k], W[k * HIDC + c], acc);
  acc = fmaxf(acc, 0.0f);
  __shared__ float red[HIDC];
  red[c] = acc; __syncthreads();
  for (int s = 64; s > 0; s >>= 1) { if (c < s) red[c] += red[c + s]; __syncthreads(); }
  const float mu = red[0] * (1.0f / HIDC);
  __syncthreads();
  const float d0 = acc - mu;
  red[c] = d0 * d0; __syncthreads();
  for (int s = 64; s > 0; s >>= 1) { if (c < s) red[c] += red[c + s]; __syncthreads(); }
  const float var = red[0] * (1.0f / HIDC);
  float v = d0 * rsqrtf(var + EPSV) * lnw[c] + lnb[c];
  v += mte[mt[n] * HIDC + c] + nte[nt[n] * HIDC + c];
  h[(size_t)n * HIDC + c] = v;
}

// ---- edge encoder pieces ---------------------------------------------------
__global__ void k_edge_mlp1(const float* __restrict__ ea, const float* __restrict__ W,
                            const float* __restrict__ b, float* __restrict__ out, int E)
{
  int i = blockIdx.x * blockDim.x + threadIdx.x;
  if (i >= E * HIDC) return;
  int e = i >> 7, c = i & 127;
  const float* a = ea + (size_t)e * 4;
  float acc = b[c];
  acc = fmaf(a[0], W[0 * HIDC + c], acc);
  acc = fmaf(a[1], W[1 * HIDC + c], acc);
  acc = fmaf(a[2], W[2 * HIDC + c], acc);
  acc = fmaf(a[3], W[3 * HIDC + c], acc);
  out[i] = fmaxf(acc, 0.0f);
}
// tecomb[t,c] = sum_k ee_te[t,k] * ee_Wc[128+k, c]   (bottom half of Wc)
__global__ void k_tecomb(const float* __restrict__ te, const float* __restrict__ Wc,
                         float* __restrict__ out)
{
  const int t = blockIdx.x, c = threadIdx.x;
  float acc = 0.0f;
  for (int k = 0; k < HIDC; ++k) acc = fmaf(te[t * HIDC + k], Wc[(HIDC + k) * HIDC + c], acc);
  out[t * HIDC + c] = acc;
}
__global__ void k_add_te(float* __restrict__ ee, const float* __restrict__ tec,
                         const int* __restrict__ et, int E)
{
  int i = blockIdx.x * blockDim.x + threadIdx.x;
  if (i >= E * HIDC) return;
  int e = i >> 7, c = i & 127;
  ee[i] += tec[et[e] * HIDC + c];
}
__global__ void k_colsum(const float* __restrict__ A, float* __restrict__ sums, int rows) {
  const int c = threadIdx.x;
  int r0 = blockIdx.x * 64, r1 = min(r0 + 64, rows);
  float s = 0.0f;
  for (int r = r0; r < r1; ++r) s += A[(size_t)r * HIDC + c];
  atomicAdd(&sums[c], s);
}
__global__ void k_fill_mean(float* __restrict__ dst, const float* __restrict__ sums,
                            float invE, int N) {
  int i = blockIdx.x * blockDim.x + threadIdx.x;
  if (i >= N * HIDC) return;
  dst[i] = sums[i & 127] * invE;
}

// ---- GATv2 per-edge passes -------------------------------------------------
__global__ void k_edge_logits(const float* __restrict__ xl, const float* __restrict__ xr,
    const float* __restrict__ ep, const int* __restrict__ src, const int* __restrict__ dst,
    const float* __restrict__ att, float* __restrict__ s, float* __restrict__ smax,
    int EA, int E)
{
  int i = blockIdx.x * blockDim.x + threadIdx.x;
  if (i >= EA * HEADS) return;
  int e = i >> 2, hd = i & 3;
  int sn = (e < E) ? src[e] : (e - E);
  int dn = (e < E) ? dst[e] : (e - E);
  const float* pl = xl + (size_t)sn * HIDC + hd * CHC;
  const float* pr = xr + (size_t)dn * HIDC + hd * CHC;
  const float* pe = ep + (size_t)e * HIDC + hd * CHC;
  const float* pa = att + hd * CHC;
  float acc = 0.0f;
#pragma unroll
  for (int c = 0; c < CHC; ++c) {
    float m = pl[c] + pr[c] + pe[c];
    m = (m > 0.0f) ? m : NEGS * m;          // LeakyReLU(0.2)
    acc = fmaf(m, pa[c], acc);
  }
  s[i] = acc;
  atomicMaxF(&smax[dn * HEADS + hd], acc);
}
__global__ void k_edge_z(const float* __restrict__ s, const float* __restrict__ smax,
    const int* __restrict__ dst, float* __restrict__ z, float* __restrict__ den,
    int EA, int E)
{
  int i = blockIdx.x * blockDim.x + threadIdx.x;
  if (i >= EA * HEADS) return;
  int e = i >> 2, hd = i & 3;
  int dn = (e < E) ? dst[e] : (e - E);
  float zv = __expf(s[i] - smax[dn * HEADS + hd]);
  z[i] = zv;
  atomicAdd(&den[dn * HEADS + hd], zv);
}
__global__ void k_edge_agg(const float* __restrict__ z, const float* __restrict__ den,
    const float* __restrict__ xl, const int* __restrict__ src, const int* __restrict__ dst,
    float* __restrict__ o, int EA, int E)
{
  int i = blockIdx.x * blockDim.x + threadIdx.x;
  if (i >= EA * HEADS) return;
  int e = i >> 2, hd = i & 3;
  int sn = (e < E) ? src[e] : (e - E);
  int dn = (e < E) ? dst[e] : (e - E);
  float a = z[i] / den[dn * HEADS + hd];
  const float* pl = xl + (size_t)sn * HIDC + hd * CHC;
  float* po = o + (size_t)dn * HIDC + hd * CHC;
#pragma unroll
  for (int c = 0; c < CHC; ++c) atomicAdd(&po[c], a * pl[c]);
}

// ---- GraphNorm -------------------------------------------------------------
__global__ void k_gm(const float* __restrict__ sums, const float* __restrict__ bo,
                     float* __restrict__ gm, int N) {
  int c = threadIdx.x;
  gm[c] = sums[c] / (float)N + bo[c];        // mean of (o + bo)
}
__global__ void k_colsq(const float* __restrict__ o, const float* __restrict__ bo,
    const float* __restrict__ ms, const float* __restrict__ gm,
    float* __restrict__ sq, int rows)
{
  const int c = threadIdx.x;
  int r0 = blockIdx.x * 64, r1 = min(r0 + 64, rows);
  const float sub = ms[c] * gm[c], bv = bo[c];
  float s = 0.0f;
  for (int r = r0; r < r1; ++r) {
    float oc = o[(size_t)r * HIDC + c] + bv - sub;
    s = fmaf(oc, oc, s);
  }
  atomicAdd(&sq[c], s);
}
__global__ void k_gnorm(const float* __restrict__ o, const float* __restrict__ bo,
    const float* __restrict__ ms, const float* __restrict__ gm, const float* __restrict__ sq,
    const float* __restrict__ gw, const float* __restrict__ gb,
    float* __restrict__ h, float* __restrict__ outslot, int N)
{
  int i = blockIdx.x * blockDim.x + threadIdx.x;
  if (i >= N * HIDC) return;
  int c = i & 127;
  float oc = o[i] + bo[c] - ms[c] * gm[c];
  float v = fmaxf(gw[c] * oc * rsqrtf(sq[c] / (float)N + EPSV) + gb[c], 0.0f);
  h[i] = v;
  outslot[i] = v;
}

// ---- skip concat + heads ---------------------------------------------------
__global__ void k_build_hcat(const float* __restrict__ outs, u16* __restrict__ hcat, int N) {
  int i = blockIdx.x * blockDim.x + threadIdx.x;
  if (i >= N * 3 * HIDC) return;
  int n = i / (3 * HIDC), f = i - n * 3 * HIDC;
  int l = f >> 7, c = f & 127;
  hcat[i] = f2bf(outs[(size_t)l * N * HIDC + (size_t)n * HIDC + c]);
}
__global__ void k_head_out(const float* __restrict__ Hin, const float* __restrict__ w,
    const float* __restrict__ b, float* __restrict__ out, int N, int K, int sigm)
{
  int n = blockIdx.x * blockDim.x + threadIdx.x;
  if (n >= N) return;
  const float* hr = Hin + (size_t)n * K;
  float acc = b[0];
  for (int k = 0; k < K; ++k) acc = fmaf(hr[k], w[k], acc);
  if (sigm) acc = 1.0f / (1.0f + __expf(-acc));
  out[n] = acc;
}

// ---------------------------------------------------------------------------
// Host launcher
// ---------------------------------------------------------------------------
static inline void gemm(const u16* A, const u16* Bt, const float* bias, float* C,
                        int M, int Ncols, int K, int relu, hipStream_t st) {
  dim3 g(Ncols / 16, (M + 63) / 64);
  k_gemm_bf16<<<g, dim3(128), 0, st>>>(A, Bt, bias, C, M, Ncols, K, relu);
}
static inline int gb(int n) { return (n + 255) / 256; }

extern "C" void kernel_launch(void* const* d_in, const int* in_sizes, int n_in,
                              void* d_out, int out_size, void* d_ws, size_t ws_size,
                              hipStream_t stream) {
  (void)n_in; (void)out_size; (void)ws_size;
  const int N  = in_sizes[0] / 8;     // x: N x 8
  const int E  = in_sizes[1] / 4;     // edge_attr: E x 4
  const int EA = E + N;               // with self loops
  const int NH = N * HIDC;

  const float* x  = (const float*)d_in[0];
  const float* ea = (const float*)d_in[1];
  // params pytree flattened with sorted keys (ASCII: capitals < lowercase):
  //  0..3  b_{W1,W2,b1,b2}   4..7  c_*   8..11 d_*
  // 12..18 ee_{W1,W2,Wc,b1,b2,bc,te}   19..22 f_*
  // 23..52 layers[l]: We,Wl,Wr,att,bl,bo,br,gn_b,gn_ms,gn_w
  // 53 ln_b 54 ln_w 55 mte 56 ne_W 57 ne_b 58 nte
  // 59..62 o_{W1,W2,b1,b2}  63 sk_W 64 sk_b
#define PP(i) ((const float*)d_in[2 + (i)])
  const int* eidx = (const int*)d_in[67];
  const int* mt   = (const int*)d_in[68];
  const int* nt   = (const int*)d_in[69];
  const int* et   = (const int*)d_in[70];
  const int* src = eidx;
  const int* dst = eidx + E;
  float* out = (float*)d_out;

  // ---- carve workspace ----
  char* wp = (char*)d_ws;
  auto carve = [&](size_t bytes) -> void* {
    void* r = (void*)wp; wp += (bytes + 255) & ~(size_t)255; return r;
  };
  float* h     = (float*)carve((size_t)NH * 4);
  u16*   h_bf  = (u16*)  carve((size_t)NH * 2);
  float* xl    = (float*)carve((size_t)NH * 4);
  float* xr    = (float*)carve((size_t)NH * 4);   // also reused as head hidden t1
  float* o     = (float*)carve((size_t)NH * 4);
  float* outs  = (float*)carve((size_t)3 * NH * 4);
  u16*   ee_bf = (u16*)  carve((size_t)EA * HIDC * 2);
  float* big   = (float*)carve((size_t)EA * HIDC * 4);  // ce_h / ce / ee_all / ep
  u16*   ebf   = (u16*)  carve((size_t)E  * HIDC * 2);  // edge bf16 temp; later hcat
  float* sbuf  = (float*)carve((size_t)EA * HEADS * 4);
  float* zbuf  = (float*)carve((size_t)EA * HEADS * 4);
  float* smax  = (float*)carve((size_t)N * HEADS * 4);
  float* den   = (float*)carve((size_t)N * HEADS * 4);
  float* sums  = (float*)carve(HIDC * 4);
  float* sqs   = (float*)carve(HIDC * 4);
  float* gmv   = (float*)carve(HIDC * 4);
  float* tec   = (float*)carve(4 * HIDC * 4);
  u16* w_eeW2 = (u16*)carve(HIDC * HIDC * 2);
  u16* w_wct  = (u16*)carve(HIDC * HIDC * 2);
  u16* w_sk   = (u16*)carve((size_t)HIDC * 3 * HIDC * 2);
  u16* w_o1   = (u16*)carve(HIDC * HIDC * 2);
  u16* w_d1   = (u16*)carve(64 * HIDC * 2);
  u16* w_c1   = (u16*)carve(64 * HIDC * 2);
  u16* w_b1   = (u16*)carve(64 * HIDC * 2);
  u16* w_f1   = (u16*)carve(32 * HIDC * 2);
  u16* w_Wl[3]; u16* w_Wr[3]; u16* w_We[3];
  for (int l = 0; l < 3; ++l) {
    w_We[l] = (u16*)carve(HIDC * HIDC * 2);
    w_Wl[l] = (u16*)carve(HIDC * HIDC * 2);
    w_Wr[l] = (u16*)carve(HIDC * HIDC * 2);
  }
  u16* hcat = ebf;  // alias: N*384 bf16 fits in E*128 bf16 region

  const int KN = HIDC * HIDC;
  // ---- weight prep: transpose + bf16 convert ----
  k_tconv<<<gb(KN), 256, 0, stream>>>(PP(13), w_eeW2, HIDC, HIDC);            // ee_W2
  k_tconv<<<gb(KN), 256, 0, stream>>>(PP(14), w_wct, HIDC, HIDC);             // ee_Wc top half
  k_tconv<<<gb(3 * KN), 256, 0, stream>>>(PP(63), w_sk, 3 * HIDC, HIDC);      // sk_W
  k_tconv<<<gb(KN), 256, 0, stream>>>(PP(59), w_o1, HIDC, HIDC);              // o_W1
  k_tconv<<<gb(64 * HIDC), 256, 0, stream>>>(PP(8),  w_d1, HIDC, 64);         // d_W1
  k_tconv<<<gb(64 * HIDC), 256, 0, stream>>>(PP(4),  w_c1, HIDC, 64);         // c_W1
  k_tconv<<<gb(64 * HIDC), 256, 0, stream>>>(PP(0),  w_b1, HIDC, 64);         // b_W1
  k_tconv<<<gb(32 * HIDC), 256, 0, stream>>>(PP(19), w_f1, HIDC, 32);         // f_W1
  for (int l = 0; l < 3; ++l) {
    const int pb = 23 + l * 10;
    k_tconv<<<gb(KN), 256, 0, stream>>>(PP(pb + 0), w_We[l], HIDC, HIDC);
    k_tconv<<<gb(KN), 256, 0, stream>>>(PP(pb + 1), w_Wl[l], HIDC, HIDC);
    k_tconv<<<gb(KN), 256, 0, stream>>>(PP(pb + 2), w_Wr[l], HIDC, HIDC);
  }
  k_tecomb<<<4, HIDC, 0, stream>>>(PP(18), PP(14), tec);  // ee_te @ ee_Wc[128:,:]

  // ---- node encoder ----
  k_node_encoder<<<N, HIDC, 0, stream>>>(x, PP(56), PP(57), PP(54), PP(53),
                                         PP(55), PP(58), mt, nt, h);

  // ---- edge encoder: ee_all (E real rows + N mean rows) ----
  const int EHF = E * HIDC;
  k_edge_mlp1<<<gb(EHF), 256, 0, stream>>>(ea, PP(12), PP(15), big, E);
  k_conv<<<gb(EHF), 256, 0, stream>>>(big, ebf, EHF);
  gemm(ebf, w_eeW2, PP(16), big, E, HIDC, HIDC, 0, stream);    // ce = ce_h @ W2 + b2
  k_conv<<<gb(EHF), 256, 0, stream>>>(big, ebf, EHF);
  gemm(ebf, w_wct, PP(17), big, E, HIDC, HIDC, 0, stream);     // ce @ Wc_top + bc
  k_add_te<<<gb(EHF), 256, 0, stream>>>(big, tec, et, E);      // + te-combined
  k_fill<<<1, HIDC, 0, stream>>>(sums, 0.0f, HIDC);
  k_colsum<<<(E + 63) / 64, HIDC, 0, stream>>>(big, sums, E);
  k_fill_mean<<<gb(NH), 256, 0, stream>>>(big + (size_t)E * HIDC, sums, 1.0f / (float)E, N);
  k_conv<<<gb(EA * HIDC), 256, 0, stream>>>(big, ee_bf, EA * HIDC);

  // ---- 3 GATv2 layers ----
  for (int l = 0; l < 3; ++l) {
    const int pb = 23 + l * 10;
    const float* att   = PP(pb + 3);
    const float* bl    = PP(pb + 4);
    const float* bo    = PP(pb + 5);
    const float* br    = PP(pb + 6);
    const float* gn_b  = PP(pb + 7);
    const float* gn_ms = PP(pb + 8);
    const float* gn_w  = PP(pb + 9);

    k_conv<<<gb(NH), 256, 0, stream>>>(h, h_bf, NH);
    gemm(h_bf, w_Wl[l], bl, xl, N, HIDC, HIDC, 0, stream);
    gemm(h_bf, w_Wr[l], br, xr, N, HIDC, HIDC, 0, stream);
    gemm(ee_bf, w_We[l], (const float*)nullptr, big, EA, HIDC, HIDC, 0, stream);  // ep

    k_fill<<<gb(N * HEADS), 256, 0, stream>>>(smax, -3.402823466e38f, N * HEADS);
    k_fill<<<gb(N * HEADS), 256, 0, stream>>>(den, 0.0f, N * HEADS);
    k_fill<<<gb(NH), 256, 0, stream>>>(o, 0.0f, NH);

    k_edge_logits<<<gb(EA * HEADS), 256, 0, stream>>>(xl, xr, big, src, dst, att,
                                                      sbuf, smax, EA, E);
    k_edge_z<<<gb(EA * HEADS), 256, 0, stream>>>(sbuf, smax, dst, zbuf, den, EA, E);
    k_edge_agg<<<gb(EA * HEADS), 256, 0, stream>>>(zbuf, den, xl, src, dst, o, EA, E);

    k_fill<<<1, HIDC, 0, stream>>>(sums, 0.0f, HIDC);
    k_fill<<<1, HIDC, 0, stream>>>(sqs, 0.0f, HIDC);
    k_colsum<<<(N + 63) / 64, HIDC, 0, stream>>>(o, sums, N);
    k_gm<<<1, HIDC, 0, stream>>>(sums, bo, gmv, N);
    k_colsq<<<(N + 63) / 64, HIDC, 0, stream>>>(o, bo, gn_ms, gmv, sqs, N);
    k_gnorm<<<gb(NH), 256, 0, stream>>>(o, bo, gn_ms, gmv, sqs, gn_w, gn_b,
                                        h, outs + (size_t)l * NH, N);
  }

  // ---- skip: h = concat(outs) @ sk_W + sk_b + h ----
  k_build_hcat<<<gb(3 * NH), 256, 0, stream>>>(outs, hcat, N);
  gemm(hcat, w_sk, PP(64), xl, N, HIDC, 3 * HIDC, 0, stream);
  k_add<<<gb(NH), 256, 0, stream>>>(h, xl, NH);

  // ---- output heads ----
  k_conv<<<gb(NH), 256, 0, stream>>>(h, h_bf, NH);
  float* t1 = xr;
  gemm(h_bf, w_o1, PP(61), t1, N, HIDC, HIDC, 1, stream);                 // order
  k_head_out<<<gb(N), 256, 0, stream>>>(t1, PP(60), PP(62), out + 0, N, HIDC, 0);
  gemm(h_bf, w_d1, PP(10), t1, N, 64, HIDC, 1, stream);                   // demand
  k_head_out<<<gb(N), 256, 0, stream>>>(t1, PP(9), PP(11), out + (size_t)N, N, 64, 0);
  gemm(h_bf, w_c1, PP(6), t1, N, 64, HIDC, 1, stream);                    // cost
  k_head_out<<<gb(N), 256, 0, stream>>>(t1, PP(5), PP(7), out + (size_t)2 * N, N, 64, 0);
  gemm(h_bf, w_b1, PP(2), t1, N, 64, HIDC, 1, stream);                    // bullwhip
  k_head_out<<<gb(N), 256, 0, stream>>>(t1, PP(1), PP(3), out + (size_t)3 * N, N, 64, 0);
  gemm(h_bf, w_f1, PP(21), t1, N, 32, HIDC, 1, stream);                   // confidence
  k_head_out<<<gb(N), 256, 0, stream>>>(t1, PP(20), PP(22), out + (size_t)4 * N, N, 32, 1);
  k_copy<<<gb(NH), 256, 0, stream>>>(out + (size_t)5 * N, h, NH);
#undef PP
}
